// cellulav2_12575664243144
// MI455X (gfx1250) — matrix operand order repaired and checked by reading, not compiled, not run on hardware
//
#include <hip/hip_runtime.h>
#include <hip/hip_bf16.h>

typedef __bf16 bf16;
typedef bf16  v16bf __attribute__((ext_vector_type(16)));
typedef float v8f   __attribute__((ext_vector_type(8)));

#define IMG_H 1024
#define IMG_W 1024
#define NBATCH 32
#define NPIX (NBATCH * IMG_H * IMG_W)  // 33554432

// ---- LDS layout for the step kernel (single shared block, halfword units) ----
// tile: 18 rows x 144 halfwords (halo col -1 at index 7, aligned cols at 8..135,
//       halo col +128 at 136). Row byte stride 288 (16-aligned).
#define T_STRIDE 144
#define TILE_HW  (18 * T_STRIDE)        // 2592 halfwords = 5184 B
#define BU_HW    (16 * 128)             // 2048 halfwords = 4096 B
#define BU_OFF   TILE_HW                // byte offset 5184 (16-aligned)
#define OUT_OFF  (TILE_HW + BU_HW)      // halfword offset 4640, byte 9280 (16-aligned)
#define SH_TOTAL (TILE_HW + BU_HW + 4096)  // + 8 KiB for the f32 out stage

// ---------------------------------------------------------------------------
// 1) per-block min/max partials (1024 blocks x 256 threads, grid-stride)
// ---------------------------------------------------------------------------
__global__ void kMinMaxPartial(const float* __restrict__ x, float* __restrict__ partials, int n) {
    __shared__ float smin[256];
    __shared__ float smax[256];
    float mn = 3.402823466e38f, mx = -3.402823466e38f;
    for (unsigned i = blockIdx.x * 256u + threadIdx.x; i < (unsigned)n; i += 256u * 1024u) {
        float v = x[i];
        mn = fminf(mn, v);
        mx = fmaxf(mx, v);
    }
    smin[threadIdx.x] = mn;
    smax[threadIdx.x] = mx;
    __syncthreads();
    for (int s = 128; s > 0; s >>= 1) {
        if ((int)threadIdx.x < s) {
            smin[threadIdx.x] = fminf(smin[threadIdx.x], smin[threadIdx.x + s]);
            smax[threadIdx.x] = fmaxf(smax[threadIdx.x], smax[threadIdx.x + s]);
        }
        __syncthreads();
    }
    if (threadIdx.x == 0) {
        partials[2 * blockIdx.x + 0] = smin[0];
        partials[2 * blockIdx.x + 1] = smax[0];
    }
}

// ---------------------------------------------------------------------------
// 2) final reduce -> norm[0] = a = 2/(max-min), norm[1] = b = -min*a - 1
// ---------------------------------------------------------------------------
__global__ void kMinMaxFinal(const float* __restrict__ partials, float* __restrict__ norm) {
    __shared__ float smin[256];
    __shared__ float smax[256];
    float mn = 3.402823466e38f, mx = -3.402823466e38f;
    for (int i = threadIdx.x; i < 1024; i += 256) {
        mn = fminf(mn, partials[2 * i + 0]);
        mx = fmaxf(mx, partials[2 * i + 1]);
    }
    smin[threadIdx.x] = mn;
    smax[threadIdx.x] = mx;
    __syncthreads();
    for (int s = 128; s > 0; s >>= 1) {
        if ((int)threadIdx.x < s) {
            smin[threadIdx.x] = fminf(smin[threadIdx.x], smin[threadIdx.x + s]);
            smax[threadIdx.x] = fmaxf(smax[threadIdx.x], smax[threadIdx.x + s]);
        }
        __syncthreads();
    }
    if (threadIdx.x == 0) {
        float a = 2.0f / (smax[0] - smin[0]);
        norm[0] = a;
        norm[1] = -smin[0] * a - 1.0f;
    }
}

// ---------------------------------------------------------------------------
// 3) drive kernel: BU = conv3x3(xn, convBx)  (bf16), and s1 = clip(BU+bias)
// ---------------------------------------------------------------------------
__global__ void __launch_bounds__(256) kDrive(const float* __restrict__ x,
                                              const float* __restrict__ wB,
                                              const float* __restrict__ bias,
                                              const float* __restrict__ norm,
                                              bf16* __restrict__ BU, bf16* __restrict__ s1) {
    __shared__ float lds[18 * 132];
    const int z = blockIdx.z;
    const int rowStart = blockIdx.y << 4;
    const int colStart = blockIdx.x << 7;
    const size_t ibase = ((size_t)z << 20);
    const float a = norm[0];
    const float b = norm[1];
    const float* img = x + ibase;

    for (int e = threadIdx.x; e < 18 * 130; e += 256) {
        int rr = e / 130, cc = e - rr * 130;
        int gy = rowStart - 1 + rr, gx = colStart - 1 + cc;
        float v = 0.0f;  // zero padding applies to normalized xn
        if ((unsigned)gy < (unsigned)IMG_H && (unsigned)gx < (unsigned)IMG_W)
            v = img[(size_t)gy * IMG_W + gx] * a + b;
        lds[rr * 132 + cc] = v;
    }
    float w[9];
#pragma unroll
    for (int i = 0; i < 9; ++i) w[i] = wB[i];
    const float bi = bias[0];
    __syncthreads();

    const int col = threadIdx.x & 127;
    const int r0 = threadIdx.x >> 7;  // 0 or 1
#pragma unroll
    for (int rr = r0; rr < 16; rr += 2) {
        const float* p = &lds[rr * 132 + col];
        float acc = w[0] * p[0] + w[1] * p[1] + w[2] * p[2];
        acc += w[3] * p[132] + w[4] * p[133] + w[5] * p[134];
        acc += w[6] * p[264] + w[7] * p[265] + w[8] * p[266];
        size_t idx = ibase + (size_t)(rowStart + rr) * IMG_W + colStart + col;
        BU[idx] = (bf16)acc;
        float s = fminf(1.0f, fmaxf(-1.0f, acc + bi));  // step 1 from s0 == 0
        s1[idx] = (bf16)s;
    }
}

// ---------------------------------------------------------------------------
// async 16B global -> LDS copy (CDNA5, ASYNCcnt-tracked)
// ---------------------------------------------------------------------------
__device__ __forceinline__ void async_b128_to_lds(unsigned ldsByteOff, unsigned long long gaddr) {
    asm volatile("global_load_async_to_lds_b128 %0, %1, off"
                 :
                 : "v"(ldsByteOff), "v"(gaddr)
                 : "memory");
}
__device__ __forceinline__ void wait_async0() {
    asm volatile("s_wait_asynccnt 0" ::: "memory");
}

// ---------------------------------------------------------------------------
// 4) step kernel: s' = clip(conv3x3(s, A) + bias + BU) with 3 accumulating
//    v_wmma_f32_16x16x32_bf16 per 16x16 tile (contract over halo rows).
//    All global traffic staged through LDS as 16B transfers.
// ---------------------------------------------------------------------------
template <int FINAL>
__global__ void __launch_bounds__(256) kStep(const bf16* __restrict__ sin,
                                             const bf16* __restrict__ BU,
                                             const float* __restrict__ wA,
                                             const float* __restrict__ bias,
                                             bf16* __restrict__ sout,
                                             float* __restrict__ fout) {
    __shared__ __align__(16) unsigned short sh[SH_TOTAL];
    const int t = threadIdx.x;
    const int z = blockIdx.z;
    const int rowStart = blockIdx.y << 4;
    const int colStart = blockIdx.x << 7;
    const size_t ibase = ((size_t)z << 20);

    // --- async: interior tile rows 1..16 (always in-bounds), 16B per thread ---
    {
        const int row = 1 + (t >> 4);         // LDS tile row 1..16  (gy = rowStart + t>>4)
        const int c8 = (t & 15) << 3;         // col (8 bf16 = 16B)
        unsigned ldsOff = (unsigned)(row * (T_STRIDE * 2) + (c8 << 1) + 16);  // halo col at idx 7 -> +16B
        unsigned long long g =
            (unsigned long long)(const void*)(sin + ibase + (size_t)(rowStart + (t >> 4)) * IMG_W + colStart + c8);
        async_b128_to_lds(ldsOff, g);
    }
    // --- async: BU tile 16x128 (always in-bounds), 16B per thread ---
    {
        const int y = t >> 4;
        const int x8 = (t & 15) << 3;
        unsigned ldsOff = (unsigned)(BU_OFF * 2 + (t << 4));
        unsigned long long g =
            (unsigned long long)(const void*)(BU + ibase + (size_t)(rowStart + y) * IMG_W + colStart + x8);
        async_b128_to_lds(ldsOff, g);
    }

    const unsigned short* sinw = (const unsigned short*)sin;
    // --- guarded halo rows 0 and 17 (256 elements, 1 per thread) ---
    {
        const int rsel = t >> 7;                       // 0 -> tile row 0, 1 -> tile row 17
        const int row = rsel ? 17 : 0;
        const int col = t & 127;
        const int gy = rowStart - 1 + row;
        unsigned short v = 0;
        if ((unsigned)gy < (unsigned)IMG_H)
            v = sinw[ibase + (size_t)gy * IMG_W + colStart + col];
        sh[row * T_STRIDE + col + 8] = v;
    }
    // --- guarded halo columns (18 rows x {-1, +128}) ---
    if (t < 36) {
        const int side = (t >= 18);                    // 0 = left (col -1), 1 = right (col 128)
        const int row = side ? (t - 18) : t;
        const int gy = rowStart - 1 + row;
        const int gx = side ? (colStart + 128) : (colStart - 1);
        unsigned short v = 0;
        if ((unsigned)gy < (unsigned)IMG_H && (unsigned)gx < (unsigned)IMG_W)
            v = sinw[ibase + (size_t)gy * IMG_W + gx];
        sh[row * T_STRIDE + (side ? 136 : 7)] = v;
    }

    // --- banded weight matrices A_dx (16-bit A layout: K = j + 8*((j>=8)+(lane>=16))) ---
    const int lane = t & 31;
    const int wv = t >> 5;       // wave id 0..7 -> tile column
    const int m = lane & 15;     // A row / B column / D column
    const int laneHi = lane >> 4;
    v16bf a0{}, a1{}, a2{};
#pragma unroll
    for (int j = 0; j < 16; ++j) {
        int K = j + 8 * ((j >> 3) + laneHi);
        int d = K - m;
        if (d >= 0 && d <= 2) {
            a0[j] = (bf16)wA[d * 3 + 0];
            a1[j] = (bf16)wA[d * 3 + 1];
            a2[j] = (bf16)wA[d * 3 + 2];
        }
    }
    const float bi = bias[0];

    wait_async0();
    __syncthreads();

    // --- C = BU + bias in D layout (VGPR r -> row r + 8*laneHi, col m) ---
    v8f c;
#pragma unroll
    for (int r = 0; r < 8; ++r) {
        unsigned u = sh[BU_OFF + ((r + (laneHi << 3)) << 7) + (wv << 4) + m];
        c[r] = __builtin_bit_cast(float, u << 16) + bi;
    }

    // --- B_dx from LDS tile: lane = column m, element j -> K = j + 8*((j>=8)+laneHi) ---
    v16bf b0{}, b1{}, b2{};
    const int cb = (wv << 4) + m + 7;  // LDS index of (x + dx - 1) for dx = 0
#pragma unroll
    for (int j = 0; j < 16; ++j) {
        int K = j + 8 * ((j >> 3) + laneHi);
        if (K < 18) {
            const unsigned short* row = &sh[K * T_STRIDE + cb];
            b0[j] = __builtin_bit_cast(bf16, row[0]);
            b1[j] = __builtin_bit_cast(bf16, row[1]);
            b2[j] = __builtin_bit_cast(bf16, row[2]);
        }
    }

    c = __builtin_amdgcn_wmma_f32_16x16x32_bf16(false, a0, false, b0, (short)0, c, false, false);
    c = __builtin_amdgcn_wmma_f32_16x16x32_bf16(false, a1, false, b1, (short)0, c, false, false);
    c = __builtin_amdgcn_wmma_f32_16x16x32_bf16(false, a2, false, b2, (short)0, c, false, false);

    // --- clip + stage result in LDS (branch-free; FINAL selected at compile time) ---
    if (FINAL) {
        float* of = (float*)&sh[OUT_OFF];
#pragma unroll
        for (int r = 0; r < 8; ++r) {
            float v = fminf(1.0f, fmaxf(-1.0f, c[r]));
            of[((r + (laneHi << 3)) << 7) + (wv << 4) + m] = (v + 1.0f) * 0.5f;
        }
    } else {
#pragma unroll
        for (int r = 0; r < 8; ++r) {
            float v = fminf(1.0f, fmaxf(-1.0f, c[r]));
            sh[OUT_OFF + ((r + (laneHi << 3)) << 7) + (wv << 4) + m] =
                __builtin_bit_cast(unsigned short, (bf16)v);
        }
    }
    __syncthreads();

    // --- coalesced write-out: 16B (bf16) or 2x16B (f32) per thread ---
    const int oy = t >> 4;
    const int ox = (t & 15) << 3;
    if (FINAL) {
        const uint4* src = (const uint4*)&sh[OUT_OFF + ((oy << 7) + ox) * 2];  // f32: *4 bytes = idx*2 halfwords
        uint4* dst = (uint4*)(fout + ibase + (size_t)(rowStart + oy) * IMG_W + colStart + ox);
        dst[0] = src[0];
        dst[1] = src[1];
    } else {
        uint4 v = *(const uint4*)&sh[OUT_OFF + (oy << 7) + ox];
        *(uint4*)((unsigned short*)sout + ibase + (size_t)(rowStart + oy) * IMG_W + colStart + ox) = v;
    }
}

// ---------------------------------------------------------------------------
// host launcher
// ---------------------------------------------------------------------------
extern "C" void kernel_launch(void* const* d_in, const int* in_sizes, int n_in,
                              void* d_out, int out_size, void* d_ws, size_t ws_size,
                              hipStream_t stream) {
    const float* x     = (const float*)d_in[0];
    const float* convA = (const float*)d_in[1];
    const float* convB = (const float*)d_in[2];
    const float* biasx = (const float*)d_in[3];
    float* out = (float*)d_out;

    // workspace layout
    float* norm     = (float*)d_ws;                 // 2 floats
    float* partials = (float*)((char*)d_ws + 256);  // 2048 floats
    bf16*  BU       = (bf16*)((char*)d_ws + 65536); // 64 MiB
    bf16*  sA       = BU + (size_t)NPIX;            // 64 MiB
    bf16*  sB       = sA + (size_t)NPIX;            // 64 MiB

    const int n = in_sizes[0];

    kMinMaxPartial<<<1024, 256, 0, stream>>>(x, partials, n);
    kMinMaxFinal<<<1, 256, 0, stream>>>(partials, norm);

    dim3 grid(IMG_W / 128, IMG_H / 16, NBATCH);
    kDrive<<<grid, 256, 0, stream>>>(x, convB, biasx, norm, BU, sA);  // BU + step 1

    const bf16* cur = sA;
    bf16* nxt = sB;
    for (int it = 0; it < 19; ++it) {  // steps 2..20
        kStep<0><<<grid, 256, 0, stream>>>(cur, BU, convA, biasx, nxt, out);
        const bf16* tp = cur;
        cur = nxt;
        nxt = (bf16*)tp;
    }
    // step 21: write (s+1)*0.5 as f32 straight to d_out
    kStep<1><<<grid, 256, 0, stream>>>(cur, BU, convA, biasx, nxt, out);
}
